// TransNAR_47132971107061
// MI455X (gfx1250) — matrix operand order repaired and checked by reading, not compiled, not run on hardware
//
#include <hip/hip_runtime.h>
#include <hip/hip_bf16.h>

// TransNAR forward for MI455X (gfx1250, wave32).
// All GEMMs run through a bf16 WMMA kernel (v_wmma_f32_16x16x32_bf16) with
// f32 accumulation. Tiles are staged to LDS with branchless, vectorized
// global b128 loads + packed f32->bf16 converts; each thread self-prefetches
// its next-K address (global_prefetch) under a uniform branch. Row pointers
// stay loop-invariant (integer K offset advances) so address-space inference
// keeps the loads on the GLOBAL path (not FLAT, which would tick DScnt and
// serialize against the WMMA fragment waits).

typedef __attribute__((ext_vector_type(16))) __bf16 bf16x16;
typedef __attribute__((ext_vector_type(4)))  __bf16 bf16x4;
typedef __attribute__((ext_vector_type(8)))  float  f32x8;
typedef __attribute__((ext_vector_type(4)))  float  f32x4;

namespace {

constexpr int kBM = 64;          // block tile M
constexpr int kBN = 128;         // block tile N
constexpr int kBK = 32;          // K step (matches WMMA K)
constexpr int kLDT = 36;         // LDS row stride in halves (72B: 8B-aligned rows,
                                 // 18-bank stride -> conflict-free 16-row frag reads)

// A fragment (16x32 bf16): lane%16 = row; lanes 0-15 hold K {0..7,16..23},
// lanes 16-31 hold K {8..15,24..31}; element h<8 -> K=8*kg+h, h>=8 -> +16.
__device__ __forceinline__ bf16x16 load_a_frag(const __bf16* tile, int row0, int lane) {
  const int r  = lane & 15;
  const int kg = (lane >> 4) & 1;
  const __bf16* p = tile + (row0 + r) * kLDT + 8 * kg;
  bf16x16 f;
#pragma unroll
  for (int h = 0; h < 8; ++h) f[h] = p[h];
#pragma unroll
  for (int h = 0; h < 8; ++h) f[8 + h] = p[16 + h];
  return f;
}

// B fragment (32x16 bf16): lane%16 = column; lanes 0-15 hold K {0..7,16..23},
// lanes 16-31 hold K {8..15,24..31}; element h<8 -> K=8*kg+h, h>=8 -> +16.
__device__ __forceinline__ bf16x16 load_b_frag(const __bf16* tile, int col0, int lane) {
  const int n  = lane & 15;
  const int kg = (lane >> 4) & 1;
  const __bf16* p = tile + (col0 + n) * kLDT + 8 * kg;
  bf16x16 f;
#pragma unroll
  for (int h = 0; h < 8; ++h) f[h] = p[h];
#pragma unroll
  for (int h = 0; h < 8; ++h) f[8 + h] = p[16 + h];
  return f;
}

// C = [relu]( A[M,K] * op(B) + bias + add ), op(B) = B[N,K]^T (TRANSB) or B[K,N].
// Batched over gridDim.z with elementwise strides sA/sB/sC/sAdd.
// Assumes: K % 32 == 0; all row strides are multiples of 4 floats (16B-aligned
// vector loads); in !TRANSB mode N % 4 == 0.
template <bool TRANSB>
__global__ __launch_bounds__(256) void gemm_bf16_wmma(
    const float* __restrict__ A, const float* __restrict__ Bm,
    const float* __restrict__ bias, const float* __restrict__ addsrc,
    float* __restrict__ C,
    int M, int N, int K, int lda, int ldb, int ldc, int ldadd,
    long long sA, long long sB, long long sC, long long sAdd, int relu)
{
  __shared__ __bf16 lA[kBM][kLDT];
  __shared__ __bf16 lB[kBN][kLDT];   // always stored as [n][k]

  const int z = blockIdx.z;
  A  += z * sA;
  Bm += z * sB;
  C  += z * sC;
  const float* addp = addsrc ? (addsrc + z * sAdd) : nullptr;

  const int tid  = threadIdx.x;
  const int lane = tid & 31;
  const int wave = tid >> 5;
  const int wm   = wave >> 2;        // 0..1 -> 32 rows each
  const int wn   = wave & 3;         // 0..3 -> 32 cols each
  const int m0   = blockIdx.y * kBM;
  const int n0   = blockIdx.x * kBN;

  const f32x8 zero8 = {0.f, 0.f, 0.f, 0.f, 0.f, 0.f, 0.f, 0.f};
  const f32x4 zero4 = {0.f, 0.f, 0.f, 0.f};
  f32x8 acc[2][2];
#pragma unroll
  for (int i = 0; i < 2; ++i)
#pragma unroll
    for (int j = 0; j < 2; ++j) acc[i][j] = zero8;

  // per-thread staging coordinates (vec4 granularity)
  const int ar0 = tid >> 3, ac0 = tid & 7;    // A/B(T): row tid/8 (+i*32), vec4 col tid%8
  const int bnv = tid & 31, bkk = tid >> 5;   // B(N):  n-vec4 tid%32, k-row tid/32 (+i*8)

  // ---- loop-invariant row pointers + bounds flags (K offset advances) ----
  const float* pa[2];
  bool aZ[2];
#pragma unroll
  for (int i = 0; i < 2; ++i) {
    int gm  = m0 + ar0 + i * 32;
    aZ[i]   = gm >= M;
    int gmc = aZ[i] ? (M - 1) : gm;
    pa[i]   = A + (long long)gmc * lda + 4 * ac0;
  }
  const float* pb[4];
  bool bZ[4];
  if (TRANSB) {
#pragma unroll
    for (int i = 0; i < 4; ++i) {
      int gn  = n0 + ar0 + i * 32;
      bZ[i]   = gn >= N;
      int gnc = bZ[i] ? (N - 1) : gn;
      pb[i]   = Bm + (long long)gnc * ldb + 4 * ac0;
    }
  } else {
    int gn  = n0 + 4 * bnv;
    bool z  = gn >= N;
    int gnc = z ? (N - 4) : gn;
#pragma unroll
    for (int i = 0; i < 4; ++i) {
      bZ[i] = z;
      pb[i] = Bm + (long long)(bkk + i * 8) * ldb + gnc;
    }
  }
  const long long bAdv = TRANSB ? (long long)kBK : (long long)kBK * ldb;
  long long ka = 0, kb = 0;                  // K offsets (floats)

  for (int k0 = 0; k0 < K; k0 += kBK) {
    // ---------- load phase: issue all global loads branchlessly ----------
    f32x4 aReg[2], bReg[4];
#pragma unroll
    for (int i = 0; i < 2; ++i) aReg[i] = *(const f32x4*)(pa[i] + ka);
#pragma unroll
    for (int i = 0; i < 4; ++i) bReg[i] = *(const f32x4*)(pb[i] + kb);

    // uniform-branch self-prefetch of next K tile (1:1 with loads)
    if (k0 + kBK < K) {
#pragma unroll
      for (int i = 0; i < 2; ++i) __builtin_prefetch(pa[i] + ka + kBK, 0, 3);
#pragma unroll
      for (int i = 0; i < 4; ++i) __builtin_prefetch(pb[i] + kb + bAdv, 0, 3);
    }

    // ---------- convert + store phase ----------
#pragma unroll
    for (int i = 0; i < 2; ++i) {
      f32x4 v = aZ[i] ? zero4 : aReg[i];
      bf16x4 h;
#pragma unroll
      for (int j = 0; j < 4; ++j) h[j] = (__bf16)v[j];
      *(bf16x4*)(&lA[ar0 + i * 32][4 * ac0]) = h;   // 8B-aligned ds_store_b64
    }
    if (TRANSB) {
#pragma unroll
      for (int i = 0; i < 4; ++i) {
        f32x4 v = bZ[i] ? zero4 : bReg[i];
        bf16x4 h;
#pragma unroll
        for (int j = 0; j < 4; ++j) h[j] = (__bf16)v[j];
        *(bf16x4*)(&lB[ar0 + i * 32][4 * ac0]) = h;
      }
    } else {
#pragma unroll
      for (int i = 0; i < 4; ++i) {
        f32x4 v = bZ[i] ? zero4 : bReg[i];
#pragma unroll
        for (int j = 0; j < 4; ++j) lB[4 * bnv + j][bkk + i * 8] = (__bf16)v[j];
      }
    }
    ka += kBK;
    kb += bAdv;
    __syncthreads();

    // ---------- WMMA compute ----------
#pragma unroll
    for (int i = 0; i < 2; ++i) {
      bf16x16 af = load_a_frag(&lA[0][0], wm * 32 + i * 16, lane);
#pragma unroll
      for (int j = 0; j < 2; ++j) {
        bf16x16 bf = load_b_frag(&lB[0][0], wn * 32 + j * 16, lane);
        acc[i][j] = __builtin_amdgcn_wmma_f32_16x16x32_bf16(
            false, af, false, bf, (short)0, acc[i][j], false, false);
      }
    }
    __syncthreads();
  }

  // C/D layout: VGPR v, lane l -> (m = v + 8*(l/16), n = l%16)
  const int nl = lane & 15;
  const int mh = (lane >> 4) * 8;
#pragma unroll
  for (int i = 0; i < 2; ++i) {
#pragma unroll
    for (int j = 0; j < 2; ++j) {
      int nn = n0 + wn * 32 + j * 16 + nl;
      if (nn >= N) continue;
#pragma unroll
      for (int v = 0; v < 8; ++v) {
        int mm = m0 + wm * 32 + i * 16 + mh + v;
        if (mm >= M) continue;
        float x = acc[i][j][v];
        if (bias) x += bias[nn];
        if (addp) x += addp[(long long)mm * ldadd + nn];
        if (relu) x = fmaxf(x, 0.f);
        C[(long long)mm * ldc + nn] = x;
      }
    }
  }
}

// ---- elementwise / reduction kernels -------------------------------------

__global__ __launch_bounds__(256) void gather_rows_kernel(
    const int* __restrict__ ids, const float* __restrict__ emb,
    float* __restrict__ out, int ntok, int E, long long total)
{
  long long idx = (long long)blockIdx.x * 256 + threadIdx.x;
  if (idx >= total) return;
  int t = (int)(idx / E);
  int j = (int)(idx % E);
  out[idx] = emb[(long long)ids[t] * E + j];
}

// in-place softmax over rows of length L, logits scaled by `scale`
__global__ __launch_bounds__(256) void softmax_rows_kernel(
    float* __restrict__ x, int L, float scale)
{
  float* p = x + (long long)blockIdx.x * L;
  __shared__ float red[256];
  const int tid = threadIdx.x;
  float m = -3.4e38f;
  for (int i = tid; i < L; i += 256) m = fmaxf(m, p[i] * scale);
  red[tid] = m; __syncthreads();
  for (int s = 128; s > 0; s >>= 1) {
    if (tid < s) red[tid] = fmaxf(red[tid], red[tid + s]);
    __syncthreads();
  }
  m = red[0]; __syncthreads();
  float sum = 0.f;
  for (int i = tid; i < L; i += 256) {
    float e = __expf(p[i] * scale - m);
    p[i] = e;
    sum += e;
  }
  red[tid] = sum; __syncthreads();
  for (int s = 128; s > 0; s >>= 1) {
    if (tid < s) red[tid] += red[tid + s];
    __syncthreads();
  }
  float inv = 1.f / red[0];
  for (int i = tid; i < L; i += 256) p[i] *= inv;
}

// out = LayerNorm(x + res) * g + b   (one block per row; out may alias x)
__global__ __launch_bounds__(256) void layernorm_res_kernel(
    const float* __restrict__ x, const float* __restrict__ res,
    const float* __restrict__ g, const float* __restrict__ b,
    float* __restrict__ out, int D)
{
  const long long row = blockIdx.x;
  const float* px = x + row * D;
  const float* pr = res + row * D;
  float* po = out + row * D;
  __shared__ float r1[256], r2[256];
  const int tid = threadIdx.x;
  float s = 0.f, ss = 0.f;
  for (int i = tid; i < D; i += 256) {
    float v = px[i] + pr[i];
    s += v; ss += v * v;
  }
  r1[tid] = s; r2[tid] = ss; __syncthreads();
  for (int st = 128; st > 0; st >>= 1) {
    if (tid < st) { r1[tid] += r1[tid + st]; r2[tid] += r2[tid + st]; }
    __syncthreads();
  }
  float mean = r1[0] / D;
  float var  = r2[0] / D - mean * mean;
  float inv  = rsqrtf(var + 1e-5f);
  for (int i = tid; i < D; i += 256)
    po[i] = (px[i] + pr[i] - mean) * inv * g[i] + b[i];
}

// per (batch, feature): top-2 values + argmax over N nodes of Bv[b, v, d]
// (consecutive threads own consecutive d -> coalesced strided scan)
__global__ __launch_bounds__(256) void top2_nodes_kernel(
    const float* __restrict__ Bv, float* __restrict__ tv, int* __restrict__ ti,
    int N, int D)
{
  int d = blockIdx.x * 256 + threadIdx.x;
  int b = blockIdx.y;
  if (d >= D) return;
  const float* p = Bv + (long long)b * N * D + d;
  float m1 = -3.4e38f, m2 = -3.4e38f;
  int i1 = 0;
  for (int v = 0; v < N; ++v) {
    float x = p[(long long)v * D];
    if (x > m1)      { m2 = m1; m1 = x; i1 = v; }
    else if (x > m2) { m2 = x; }
  }
  tv[((long long)b * D + d) * 2 + 0] = m1;
  tv[((long long)b * D + d) * 2 + 1] = m2;
  ti[(long long)b * D + d] = i1;
}

// out[b,u,d] = A[b,u,d] + (argmax==u ? second : max)
__global__ __launch_bounds__(256) void max_message_kernel(
    const float* __restrict__ Aarr, const float* __restrict__ tv,
    const int* __restrict__ ti, float* __restrict__ out,
    int N, int D, long long total)
{
  long long idx = (long long)blockIdx.x * 256 + threadIdx.x;
  if (idx >= total) return;
  int d = (int)(idx % D);
  long long t = idx / D;
  int u = (int)(t % N);
  int b = (int)(t / N);
  long long fd = (long long)b * D + d;
  float m1 = tv[fd * 2 + 0];
  float m2 = tv[fd * 2 + 1];
  int   i1 = ti[fd];
  out[idx] = Aarr[idx] + ((i1 == u) ? m2 : m1);
}

// ---- host-side launch helper ---------------------------------------------

inline void gemm(hipStream_t s, bool transb,
                 const float* A, const float* B, const float* bias,
                 const float* add, float* C,
                 int M, int N, int K, int lda, int ldb, int ldc, int ldadd,
                 long long sA, long long sB, long long sC, long long sAdd,
                 int batch, int relu)
{
  dim3 grid((N + kBN - 1) / kBN, (M + kBM - 1) / kBM, batch);
  if (transb)
    gemm_bf16_wmma<true><<<grid, 256, 0, s>>>(A, B, bias, add, C, M, N, K,
        lda, ldb, ldc, ldadd, sA, sB, sC, sAdd, relu);
  else
    gemm_bf16_wmma<false><<<grid, 256, 0, s>>>(A, B, bias, add, C, M, N, K,
        lda, ldb, ldc, ldadd, sA, sB, sC, sAdd, relu);
}

} // namespace

extern "C" void kernel_launch(void* const* d_in, const int* in_sizes, int n_in,
                              void* d_out, int out_size, void* d_ws, size_t ws_size,
                              hipStream_t stream)
{
  (void)in_sizes; (void)n_in; (void)out_size; (void)ws_size;

  constexpr int B = 4, S = 512, NG = 8192, DM = 768, P = 512, DK = 32,
                FF = 2048, VOCAB = 30522, NH = 8, HD = 64, L = 2;
  constexpr int TT = B * S;       // 2048 text tokens
  constexpr int GT = B * NG;      // 32768 graph nodes

  const int*   ids   = (const int*)d_in[0];
  const float* graph = (const float*)d_in[1];
  auto pf = [&](int i) { return (const float*)d_in[i]; };
  const float* emb    = pf(2);
  const float* proj_w = pf(3);  const float* proj_b = pf(4);
  const float* q_w    = pf(5);  const float* q_b    = pf(6);
  const float* k_w    = pf(7);  const float* k_b    = pf(8);
  const float* v_w    = pf(9);  const float* v_b    = pf(10);
  const float* ffn_w  = pf(11); const float* ffn_b  = pf(12);
  const float* gen_w  = pf(13); const float* gen_b  = pf(14);
  constexpr int TB = 15;   // t_layers: 12 tensors per layer
  constexpr int MB = 39;   // m_layers: 4 tensors per layer

  // workspace carve-out (floats)
  float* ws = (float*)d_ws;
  size_t off = 0;
  auto wsalloc = [&](size_t n) { float* p = ws + off; off += n; return p; };
  float* G    = wsalloc((size_t)GT * P);   // persistent graph state
  float* T    = wsalloc((size_t)TT * P);   // persistent text state
  float* buf1 = wsalloc((size_t)GT * P);   // gather / qkv / ffn-hidden / A / H1
  float* buf2 = wsalloc((size_t)GT * P);   // mha scores / Bv / V
  float* buf3 = wsalloc((size_t)GT * P);   // max_msg / cross scores
  float* bufP = wsalloc((size_t)TT * P);   // per-token temp
  float* bufO = wsalloc((size_t)TT * P);   // attention outputs
  float* bufK = wsalloc((size_t)GT * DK);
  float* bufQ = wsalloc((size_t)TT * DK);
  float* tv   = wsalloc((size_t)B * P * 2);
  int*   ti   = (int*)wsalloc((size_t)B * P);

  // ---- input projections ----
  {
    long long tot = (long long)TT * DM;
    gather_rows_kernel<<<dim3((unsigned)((tot + 255) / 256)), 256, 0, stream>>>(
        ids, emb, buf1, TT, DM, tot);
  }
  gemm(stream, true, buf1,  proj_w, proj_b, nullptr, T, TT, P, DM, DM, DM, P, 0, 0,0,0,0, 1, 0);
  gemm(stream, true, graph, proj_w, proj_b, nullptr, G, GT, P, DM, DM, DM, P, 0, 0,0,0,0, 1, 0);

  for (int l = 0; l < L; ++l) {
    const float* in_w  = pf(TB + 12*l + 0); const float* in_b  = pf(TB + 12*l + 1);
    const float* out_w = pf(TB + 12*l + 2); const float* out_b = pf(TB + 12*l + 3);
    const float* ln1_g = pf(TB + 12*l + 4); const float* ln1_b = pf(TB + 12*l + 5);
    const float* ln2_g = pf(TB + 12*l + 6); const float* ln2_b = pf(TB + 12*l + 7);
    const float* ff1_w = pf(TB + 12*l + 8); const float* ff1_b = pf(TB + 12*l + 9);
    const float* ff2_w = pf(TB + 12*l +10); const float* ff2_b = pf(TB + 12*l +11);
    const float* msg_w = pf(MB + 4*l + 0);  const float* msg_b = pf(MB + 4*l + 1);
    const float* upd_w = pf(MB + 4*l + 2);  const float* upd_b = pf(MB + 4*l + 3);

    // ================= transformer encoder layer on T =================
    // qkv = T @ in_w^T + in_b : [TT, 3P]
    gemm(stream, true, T, in_w, in_b, nullptr, buf1, TT, 3*P, P, P, P, 3*P, 0, 0,0,0,0, 1, 0);
    // scores[b,h] = q_h @ k_h^T  (layout buf2: [b][h][S][S])
    for (int h = 0; h < NH; ++h) {
      gemm(stream, true,
           buf1 + h*HD,                 // q_h   (lda = 3P)
           buf1 + P + h*HD,             // k_h   (ldb = 3P)
           nullptr, nullptr,
           buf2 + (size_t)h * S * S,
           S, S, HD, 3*P, 3*P, S, 0,
           (long long)S * 3*P, (long long)S * 3*P, (long long)NH * S * S, 0,
           B, 0);
    }
    softmax_rows_kernel<<<dim3(B * NH * S), 256, 0, stream>>>(buf2, S, 0.125f); // 1/sqrt(64)
    // o_h = scores @ v_h  -> bufO[b, s, h*HD + d]
    for (int h = 0; h < NH; ++h) {
      gemm(stream, false,
           buf2 + (size_t)h * S * S,    // attn  (lda = S)
           buf1 + 2*P + h*HD,           // v_h as B[K=S, N=HD], ldb = 3P
           nullptr, nullptr,
           bufO + h*HD,
           S, HD, S, S, 3*P, P, 0,
           (long long)NH * S * S, (long long)S * 3*P, (long long)S * P, 0,
           B, 0);
    }
    gemm(stream, true, bufO, out_w, out_b, nullptr, bufP, TT, P, P, P, P, P, 0, 0,0,0,0, 1, 0);
    layernorm_res_kernel<<<dim3(TT), 256, 0, stream>>>(T, bufP, ln1_g, ln1_b, T, P);
    gemm(stream, true, T,    ff1_w, ff1_b, nullptr, buf1, TT, FF, P, P,  P,  FF, 0, 0,0,0,0, 1, 1); // relu
    gemm(stream, true, buf1, ff2_w, ff2_b, nullptr, bufP, TT, P,  FF, FF, FF, P,  0, 0,0,0,0, 1, 0);
    layernorm_res_kernel<<<dim3(TT), 256, 0, stream>>>(T, bufP, ln2_g, ln2_b, T, P);

    // ================= max-MPNN on G =================
    gemm(stream, true, G, msg_w,     msg_b,   nullptr, buf1, GT, P, P, P, 2*P, P, 0, 0,0,0,0, 1, 0); // A
    gemm(stream, true, G, msg_w + P, nullptr, nullptr, buf2, GT, P, P, P, 2*P, P, 0, 0,0,0,0, 1, 0); // Bv
    top2_nodes_kernel<<<dim3((P + 255) / 256, B), 256, 0, stream>>>(buf2, tv, ti, NG, P);
    {
      long long tot = (long long)GT * P;
      max_message_kernel<<<dim3((unsigned)((tot + 255) / 256)), 256, 0, stream>>>(
          buf1, tv, ti, buf3, NG, P, tot);
    }
    gemm(stream, true, G,    upd_w,     nullptr, nullptr, buf1, GT, P, P, P, 2*P, P, 0, 0,0,0,0, 1, 0); // H1
    gemm(stream, true, buf3, upd_w + P, upd_b,   buf1,    G,    GT, P, P, P, 2*P, P, P, 0,0,0,0, 1, 0); // G = H1 + ...

    // ================= cross attention T <- G =================
    gemm(stream, true, T, q_w, q_b, nullptr, bufQ, TT, DK, P, P, P, DK, 0, 0,0,0,0, 1, 0);
    gemm(stream, true, G, k_w, k_b, nullptr, bufK, GT, DK, P, P, P, DK, 0, 0,0,0,0, 1, 0);
    gemm(stream, true, G, v_w, v_b, nullptr, buf2, GT, P,  P, P, P, P,  0, 0,0,0,0, 1, 0);
    // scores[b] = Q_b @ K_b^T : [S, NG]
    gemm(stream, true, bufQ, bufK, nullptr, nullptr, buf3,
         S, NG, DK, DK, DK, NG, 0,
         (long long)S * DK, (long long)NG * DK, (long long)S * NG, 0, B, 0);
    softmax_rows_kernel<<<dim3(B * S), 256, 0, stream>>>(buf3, NG, 0.1767766953f); // 1/sqrt(32)
    // O[b] = attn @ V_b : [S, P]
    gemm(stream, false, buf3, buf2, nullptr, nullptr, bufO,
         S, P, NG, NG, P, P, 0,
         (long long)S * NG, (long long)NG * P, (long long)S * P, 0, B, 0);
    gemm(stream, true, bufO, ffn_w, ffn_b, nullptr, T, TT, P, P, P, P, P, 0, 0,0,0,0, 1, 0);
  }

  // ---- generator: logits = T @ gen_w^T + gen_b : [TT, VOCAB] ----
  gemm(stream, true, T, gen_w, gen_b, nullptr, (float*)d_out,
       TT, VOCAB, P, P, P, VOCAB, 0, 0,0,0,0, 1, 0);
}
